// aggregator_event_PECF_50345606643999
// MI455X (gfx1250) — matrix operand order repaired and checked by reading, not compiled, not run on hardware
//
#include <hip/hip_runtime.h>
#include <stdint.h>

// ---------------- problem constants ----------------
#define NNODES   50000
#define NRELS    256
#define DIM      256
#define TT       24
#define EE       20000
#define BB       16
#define SEQ      7

// ---------------- vector types ----------------
typedef __attribute__((ext_vector_type(16))) __bf16 v16bf;
typedef __attribute__((ext_vector_type(8)))  float  v8f;
typedef __attribute__((ext_vector_type(4)))  int    i4;
typedef __attribute__((ext_vector_type(4)))  float  f4;

union Frag {
    v16bf v;
    i4    i2[2];
    unsigned short s[16];
};

static __device__ __forceinline__ unsigned short f2bf(float f) {
    unsigned int u = __float_as_uint(f);
    unsigned int r = u + 0x7FFFu + ((u >> 16) & 1u);
    return (unsigned short)(r >> 16);
}

static __device__ __forceinline__ v8f wmma_bf16(const Frag& a, const Frag& b, v8f c) {
    return __builtin_amdgcn_wmma_f32_16x16x32_bf16(false, a.v, false, b.v,
                                                   (short)0, c, false, false);
}

static __device__ __forceinline__ Frag loadB(const unsigned int* p) {
    Frag f;
    f.i2[0] = ((const i4*)p)[0];
    f.i2[1] = ((const i4*)p)[1];
    return f;
}

static __device__ __forceinline__ float sigmoidf_(float x) {
    return 1.0f / (1.0f + expf(-x));
}

// ---------------- workspace layout (bytes) ----------------
#define OFF_NODEBF  0u                       // 50000*256*2   = 25,600,000
#define OFF_WBIG    25600000u                // 768*256*4     =    786,432
#define OFF_BC      26386432u                // 256*4 (pad)   =      1,024
#define OFF_PACKE   26387456u                // 98304*4       =    393,216
#define OFF_PACKIH  26780672u                //               =    393,216
#define OFF_PACKHH  27173888u                //               =    393,216
#define OFF_SUMS    27567104u                // 24*256*256*4  =  6,291,456
#define OFF_CNT     33858560u                // 24*256*4      =     24,576
#define OFF_RELBF   33883136u                // 24*256*256*2  =  3,145,728
// total ≈ 37,028,864 bytes

// ---------------- prep kernels ----------------

__global__ void k_node_bf16(const float* __restrict__ node, unsigned short* __restrict__ out) {
    int idx = blockIdx.x * blockDim.x + threadIdx.x;
    if (idx < NNODES * DIM) out[idx] = f2bf(node[idx]);
}

// Wc = text_w_W @ edge_w_W[256:512]  -> Wbig rows 256..511
__global__ void k_wc(const float* __restrict__ textW, const float* __restrict__ edgeW,
                     float* __restrict__ wbig) {
    int i = blockIdx.x;        // 0..255  (k of text input)
    int j = threadIdx.x;       // 0..255  (output col)
    float acc = 0.f;
    for (int d = 0; d < DIM; d++)
        acc += textW[i * DIM + d] * edgeW[(256 + d) * DIM + j];
    wbig[(256 + i) * DIM + j] = acc;
}

// copy rows 0..255 and 512..767 of edge_w_W into Wbig
__global__ void k_copyrows(const float* __restrict__ edgeW, float* __restrict__ wbig) {
    int idx = blockIdx.x * blockDim.x + threadIdx.x;   // < 512*256
    if (idx >= 512 * DIM) return;
    int rsel = idx >> 8;
    int col  = idx & 255;
    int row  = (rsel < 256) ? rsel : (rsel + 256);
    wbig[row * DIM + col] = edgeW[row * DIM + col];
}

// bc = text_w_b @ edge_w_W[256:512] + edge_w_b
__global__ void k_bc(const float* __restrict__ textB, const float* __restrict__ edgeW,
                     const float* __restrict__ edgeB, float* __restrict__ bc) {
    int j = threadIdx.x;
    float acc = edgeB[j];
    for (int d = 0; d < DIM; d++)
        acc += textB[d] * edgeW[(256 + d) * DIM + j];
    bc[j] = acc;
}

// pack row-major f32 W[K][N] into wave32 WMMA B-fragment dword layout.
// dword idx = ((kc*NT + nt)*32 + lane)*8 + v
//   col = nt*16 + (lane&15); k0 = kc*32 + (lane<16?0:8) + 2v + (v>=4?8:0)
__global__ void k_pack(const float* __restrict__ W, int K, int N, unsigned int* __restrict__ packed) {
    int idx = blockIdx.x * blockDim.x + threadIdx.x;
    int total = (K >> 5) * (N >> 4) * 256;
    if (idx >= total) return;
    int v    = idx & 7;
    int lane = (idx >> 3) & 31;
    int tile = idx >> 8;
    int NT   = N >> 4;
    int nt   = tile % NT;
    int kc   = tile / NT;
    int col  = nt * 16 + (lane & 15);
    int k0   = kc * 32 + ((lane < 16) ? 0 : 8) + 2 * v + ((v >= 4) ? 8 : 0);
    unsigned int lo = f2bf(W[k0 * N + col]);
    unsigned int hi = f2bf(W[(k0 + 1) * N + col]);
    packed[idx] = lo | (hi << 16);
}

__global__ void k_zero(float* __restrict__ p, int n) {
    int idx = blockIdx.x * blockDim.x + threadIdx.x;
    if (idx < n) p[idx] = 0.f;
}

// ---------------- fused edge MLP + scatter ----------------
// One wave handles 16 edges of timestep t:
//   O[16,256] = relu( [Hsrc | Text | Hdst] (16x768, bf16) @ Wbig_packed + bc )
//   atomic-add rows into sums[t, rel, :], counts[t, rel] += 1
__global__ __launch_bounds__(128) void k_edge(
        const float* __restrict__ text_emb,
        const int* __restrict__ esrc, const int* __restrict__ edst,
        const int* __restrict__ erel,
        const unsigned short* __restrict__ node_bf,
        const unsigned int* __restrict__ packE,
        const float* __restrict__ bc,
        float* __restrict__ sums, float* __restrict__ counts) {
    const int wave = threadIdx.x >> 5;
    const int lane = threadIdx.x & 31;
    const int blocksPerT = (EE + 63) / 64;          // 313
    const int t  = blockIdx.x / blocksPerT;
    const int eb = (blockIdx.x % blocksPerT) * 64 + wave * 16;
    if (eb >= EE) return;

    const int row  = lane & 15;
    const int hi   = lane >> 4;                     // 0 or 1
    const int s    = hi * 8;
    const int col0 = lane & 15;
    const long teBase = (long)t * EE;
    const int  e   = eb + row;

    const int sidx = esrc[teBase + e];
    const int didx = edst[teBase + e];
    const unsigned short* srcp = node_bf + (size_t)sidx * DIM;
    const unsigned short* dstp = node_bf + (size_t)didx * DIM;
    const float*          txtp = text_emb + (size_t)(teBase + e) * DIM;

    if (lane < 16) atomicAdd(&counts[t * NRELS + erel[teBase + e]], 1.0f);

    v8f acc[16];
    const v8f Z = {0.f,0.f,0.f,0.f,0.f,0.f,0.f,0.f};
#pragma unroll
    for (int nt = 0; nt < 16; nt++) acc[nt] = Z;

#pragma unroll
    for (int kc = 0; kc < 24; kc++) {
        Frag a;
        const int kl = (kc & 7) * 32 + s;           // element offset within 256-wide region
        if (kc < 8) {                               // h_src (bf16)
            a.i2[0] = *(const i4*)(srcp + kl);
            a.i2[1] = *(const i4*)(srcp + kl + 16);
        } else if (kc < 16) {                       // text (f32 -> bf16 in-register)
            const float* p = txtp + kl;
            f4 f0 = ((const f4*)p)[0];
            f4 f1 = ((const f4*)p)[1];
            f4 f2 = *(const f4*)(p + 16);
            f4 f3 = *(const f4*)(p + 20);
            a.s[0]=f2bf(f0.x); a.s[1]=f2bf(f0.y); a.s[2]=f2bf(f0.z); a.s[3]=f2bf(f0.w);
            a.s[4]=f2bf(f1.x); a.s[5]=f2bf(f1.y); a.s[6]=f2bf(f1.z); a.s[7]=f2bf(f1.w);
            a.s[8]=f2bf(f2.x); a.s[9]=f2bf(f2.y); a.s[10]=f2bf(f2.z); a.s[11]=f2bf(f2.w);
            a.s[12]=f2bf(f3.x); a.s[13]=f2bf(f3.y); a.s[14]=f2bf(f3.z); a.s[15]=f2bf(f3.w);
        } else {                                    // h_dst (bf16)
            a.i2[0] = *(const i4*)(dstp + kl);
            a.i2[1] = *(const i4*)(dstp + kl + 16);
        }
#pragma unroll
        for (int nt = 0; nt < 16; nt++) {
            Frag b = loadB(packE + (((size_t)kc * 16 + nt) * 32 + lane) * 8);
            acc[nt] = wmma_bf16(a, b, acc[nt]);
        }
    }

    // bias + relu + scatter
    float bcv[16];
#pragma unroll
    for (int nt = 0; nt < 16; nt++) bcv[nt] = bc[nt * 16 + col0];

#pragma unroll
    for (int r = 0; r < 8; r++) {
        const int m = r + hi * 8;
        const int relm = erel[teBase + eb + m];
        float* srow = sums + ((size_t)t * NRELS + relm) * DIM;
#pragma unroll
        for (int nt = 0; nt < 16; nt++) {
            float v = acc[nt][r] + bcv[nt];
            v = v > 0.f ? v : 0.f;
            atomicAdd(srow + nt * 16 + col0, v);
        }
    }
}

// ---------------- scatter-mean finalize + bf16 cast ----------------
__global__ void k_rel(const float* __restrict__ sums, const float* __restrict__ counts,
                      unsigned short* __restrict__ relbf) {
    int idx = blockIdx.x * blockDim.x + threadIdx.x;
    if (idx >= TT * NRELS * DIM) return;
    float c = counts[idx >> 8];
    float v = (c > 0.5f) ? (sums[idx] / c) : 0.f;
    relbf[idx] = f2bf(v);
}

// ---------------- GRU over 7 steps ----------------
// Block = 128 threads = 4 waves; wave handles 16 relation rows of batch b.
__global__ __launch_bounds__(128) void k_gru(
        const unsigned short* __restrict__ relbf,
        const int* __restrict__ time_idx,
        const unsigned int* __restrict__ packIH,
        const unsigned int* __restrict__ packHH,
        const float* __restrict__ bih, const float* __restrict__ bhh,
        float* __restrict__ out) {
    __shared__ __align__(16) unsigned short lds[4][16 * DIM];   // bf16 h per wave (8KB each)

    const int wave = threadIdx.x >> 5;
    const int lane = threadIdx.x & 31;
    const int b    = blockIdx.x >> 2;
    const int r0   = (blockIdx.x & 3) * 64 + wave * 16;
    const int row  = lane & 15;
    const int hi   = lane >> 4;
    const int s    = hi * 8;
    const int col0 = lane & 15;
    unsigned short* ldsw = lds[wave];

    // zero h in LDS (4096 halfs = 512 i4 / 32 lanes = 16 per lane)
    {
        i4 z = {0,0,0,0};
#pragma unroll
        for (int i = 0; i < 16; i++) ((i4*)ldsw)[lane + i * 32] = z;
    }
    asm volatile("s_wait_dscnt 0" ::: "memory");

    const v8f Z = {0.f,0.f,0.f,0.f,0.f,0.f,0.f,0.f};
    v8f h[16];
#pragma unroll
    for (int nt = 0; nt < 16; nt++) h[nt] = Z;

    for (int step = 0; step < SEQ; step++) {
        const int t = time_idx[b * SEQ + step];
        const unsigned short* xrow = relbf + ((size_t)t * NRELS + r0 + row) * DIM;
        const unsigned short* hrow = ldsw + row * DIM;

        Frag xf[8], hf[8];
#pragma unroll
        for (int kc = 0; kc < 8; kc++) {
            const int kl = kc * 32 + s;
            xf[kc].i2[0] = *(const i4*)(xrow + kl);
            xf[kc].i2[1] = *(const i4*)(xrow + kl + 16);
            hf[kc].i2[0] = *(const i4*)(hrow + kl);
            hf[kc].i2[1] = *(const i4*)(hrow + kl + 16);
        }

#pragma unroll
        for (int nt = 0; nt < 16; nt++) {
            v8f air = Z, aiz = Z, ain = Z, ahr = Z, ahz = Z, ahn = Z;
#pragma unroll
            for (int kc = 0; kc < 8; kc++) {
                const size_t kb = (size_t)kc * 48;
                Frag b0 = loadB(packIH + ((kb + nt     ) * 32 + lane) * 8);
                air = wmma_bf16(xf[kc], b0, air);
                Frag b1 = loadB(packIH + ((kb + 16 + nt) * 32 + lane) * 8);
                aiz = wmma_bf16(xf[kc], b1, aiz);
                Frag b2 = loadB(packIH + ((kb + 32 + nt) * 32 + lane) * 8);
                ain = wmma_bf16(xf[kc], b2, ain);
                Frag b3 = loadB(packHH + ((kb + nt     ) * 32 + lane) * 8);
                ahr = wmma_bf16(hf[kc], b3, ahr);
                Frag b4 = loadB(packHH + ((kb + 16 + nt) * 32 + lane) * 8);
                ahz = wmma_bf16(hf[kc], b4, ahz);
                Frag b5 = loadB(packHH + ((kb + 32 + nt) * 32 + lane) * 8);
                ahn = wmma_bf16(hf[kc], b5, ahn);
            }
            const int c = nt * 16 + col0;
            const float br = bih[c],       bz = bih[256 + c], bn = bih[512 + c];
            const float cr = bhh[c],       cz = bhh[256 + c], cn = bhh[512 + c];
            v8f hnew;
#pragma unroll
            for (int r = 0; r < 8; r++) {
                float rg = sigmoidf_(air[r] + br + ahr[r] + cr);
                float zg = sigmoidf_(aiz[r] + bz + ahz[r] + cz);
                float ng = tanhf(ain[r] + bn + rg * (ahn[r] + cn));
                hnew[r] = (1.f - zg) * ng + zg * h[nt][r];
            }
            h[nt] = hnew;
#pragma unroll
            for (int r = 0; r < 8; r++) {
                const int m = r + hi * 8;
                ldsw[m * DIM + c] = f2bf(hnew[r]);
            }
        }
        asm volatile("s_wait_dscnt 0" ::: "memory");
    }

    // write final hidden state: out[b, r, d] (f32)
#pragma unroll
    for (int nt = 0; nt < 16; nt++) {
#pragma unroll
        for (int r = 0; r < 8; r++) {
            const int m = r + hi * 8;
            out[((size_t)(b * NRELS + r0 + m)) * DIM + nt * 16 + col0] = h[nt][r];
        }
    }
}

// ---------------- launcher ----------------
extern "C" void kernel_launch(void* const* d_in, const int* in_sizes, int n_in,
                              void* d_out, int out_size, void* d_ws, size_t ws_size,
                              hipStream_t stream) {
    const float* node     = (const float*)d_in[0];
    const float* text_emb = (const float*)d_in[1];
    const float* textW    = (const float*)d_in[2];
    const float* textB    = (const float*)d_in[3];
    const float* edgeW    = (const float*)d_in[4];
    const float* edgeB    = (const float*)d_in[5];
    const float* gruWih   = (const float*)d_in[6];
    const float* gruWhh   = (const float*)d_in[7];
    const float* gruBih   = (const float*)d_in[8];
    const float* gruBhh   = (const float*)d_in[9];
    const int*   esrc     = (const int*)d_in[10];
    const int*   edst     = (const int*)d_in[11];
    const int*   erel     = (const int*)d_in[12];
    const int*   tidx     = (const int*)d_in[13];
    float* out = (float*)d_out;

    char* ws = (char*)d_ws;
    unsigned short* node_bf = (unsigned short*)(ws + OFF_NODEBF);
    float*          wbig    = (float*)(ws + OFF_WBIG);
    float*          bc      = (float*)(ws + OFF_BC);
    unsigned int*   packE   = (unsigned int*)(ws + OFF_PACKE);
    unsigned int*   packIH  = (unsigned int*)(ws + OFF_PACKIH);
    unsigned int*   packHH  = (unsigned int*)(ws + OFF_PACKHH);
    float*          sums    = (float*)(ws + OFF_SUMS);
    float*          counts  = (float*)(ws + OFF_CNT);
    unsigned short* relbf   = (unsigned short*)(ws + OFF_RELBF);

    // --- prep ---
    k_node_bf16<<<(NNODES * DIM + 255) / 256, 256, 0, stream>>>(node, node_bf);
    k_wc<<<256, 256, 0, stream>>>(textW, edgeW, wbig);
    k_copyrows<<<512, 256, 0, stream>>>(edgeW, wbig);
    k_bc<<<1, 256, 0, stream>>>(textB, edgeW, edgeB, bc);
    k_pack<<<384, 256, 0, stream>>>(wbig,   768, 256, packE);
    k_pack<<<384, 256, 0, stream>>>(gruWih, 256, 768, packIH);
    k_pack<<<384, 256, 0, stream>>>(gruWhh, 256, 768, packHH);

    // zero sums + counts (contiguous region)
    const int nzero = (TT * NRELS * DIM) + (TT * NRELS);
    k_zero<<<(nzero + 255) / 256, 256, 0, stream>>>(sums, nzero);

    // --- fused edge MLP + scatter-sum ---
    const int blocksPerT = (EE + 63) / 64;      // 313
    k_edge<<<TT * blocksPerT, 128, 0, stream>>>(text_emb, esrc, edst, erel,
                                                node_bf, packE, bc, sums, counts);

    // --- mean + bf16 ---
    k_rel<<<(TT * NRELS * DIM + 255) / 256, 256, 0, stream>>>(sums, counts, relbf);

    // --- GRU ---
    k_gru<<<BB * 4, 128, 0, stream>>>(relbf, tidx, packIH, packHH, gruBih, gruBhh, out);
}
